// EuclideanCodebook_9706626089876
// MI455X (gfx1250) — compile-verified, matrix operand order checked by reading, not run on hardware
//
#include <hip/hip_runtime.h>
#include <hip/hip_bf16.h>

typedef __bf16 bf16_t;
typedef __attribute__((ext_vector_type(16))) __bf16 v16bf;
typedef __attribute__((ext_vector_type(8)))  __bf16 v8bf;
typedef __attribute__((ext_vector_type(8)))  float  v8f;

#define N_ROWS   32768   // 8*4096 flattened x rows
#define N_CODES  8192
#define DIM      256
#define K_STEPS  8       // 256 / 32
#define AST      264     // padded LDS row stride (bf16 elems): 528B bank-stagger
#define M_TILES  2       // 2 x 16 = 32 rows per block (fits A in VGPRs, no spill)
#define BLK_ROWS (M_TILES * 16)

// workspace layout (bytes)
#define WS_BHI_OFF  0
#define WS_BLO_OFF  ((size_t)N_CODES * DIM * 2)   // 4 MiB
#define WS_BIAS_OFF ((size_t)N_CODES * DIM * 4)   // 8 MiB

// ---------------------------------------------------------------------------
// bias_c = -0.5 * ||embed_c||^2
__global__ void vq_bias_kernel(const float* __restrict__ embed,
                               float* __restrict__ bias) {
    int c = blockIdx.x * blockDim.x + threadIdx.x;
    if (c >= N_CODES) return;
    const float* e = embed + (size_t)c * DIM;
    float s = 0.f;
    #pragma unroll 4
    for (int k = 0; k < DIM; ++k) { float v = e[k]; s += v * v; }
    bias[c] = -0.5f * s;
}

// ---------------------------------------------------------------------------
// Pack embed into bf16 hi/lo with the ISA 16-bit B-operand layout:
// for code-tile ct, k-step s, lane L: element j holds
//   K = s*32 + (L/16)*8 + (j<8 ? j : 8+j),  column n = ct*16 + L%16.
__global__ void vq_pack_kernel(const float* __restrict__ embed,
                               bf16_t* __restrict__ bhi,
                               bf16_t* __restrict__ blo) {
    int g = blockIdx.x * blockDim.x + threadIdx.x;   // 0 .. 131071
    int ct   = g >> 8;
    int rem  = g & 255;
    int s    = rem >> 5;
    int lane = rem & 31;
    int n     = ct * 16 + (lane & 15);
    int kbase = s * 32 + (lane >> 4) * 8;
    size_t out = (size_t)g * 16;                     // == ((ct*8+s)*32+lane)*16
    #pragma unroll
    for (int j = 0; j < 16; ++j) {
        int k = kbase + (j < 8 ? j : 8 + j);         // j>=8 -> kbase+16+(j-8)
        float v = embed[(size_t)n * DIM + k];
        bf16_t h = (bf16_t)v;
        bhi[out + j] = h;
        blo[out + j] = (bf16_t)(v - (float)h);
    }
}

// ---------------------------------------------------------------------------
// Fused GEMM + row-argmax. One block = 32 x-rows (2 M-tiles), 8 waves; each
// wave owns one 16-column tile per sweep step. A stays register-resident
// (2 tiles * 8 ksteps * hi/lo = 256 VGPRs, no spill); each packed-B read is
// amortized over both M-tiles (halves L2 traffic vs 16-row blocks).
__global__ __launch_bounds__(256) void vq_argmin_kernel(
    const float*  __restrict__ x,
    const float*  __restrict__ embed,
    const bf16_t* __restrict__ bhi,
    const bf16_t* __restrict__ blo,
    const float*  __restrict__ bias,
    float* __restrict__ out_q,
    float* __restrict__ out_i)
{
    __shared__ __align__(32) bf16_t sAhi[BLK_ROWS * AST];
    __shared__ __align__(32) bf16_t sAlo[BLK_ROWS * AST];
    __shared__ float r_val[8][BLK_ROWS][16];
    __shared__ int   r_idx[8][BLK_ROWS][16];
    __shared__ int   s_best[BLK_ROWS];

    const int t    = threadIdx.x;
    const int row0 = blockIdx.x * BLK_ROWS;

    // Stage 32 x-rows to LDS as bf16 hi/lo. 8192 elems / 256 thr = 32 each.
    {
        int row8 = t >> 3;             // 0..31 (one row per 8 threads)
        int c0   = (t & 7) * 32;       // 32 consecutive cols per thread
        const float* src = x + (size_t)(row0 + row8) * DIM + c0;
        bf16_t* dh = sAhi + row8 * AST + c0;
        bf16_t* dl = sAlo + row8 * AST + c0;
        #pragma unroll 8
        for (int i = 0; i < 32; ++i) {
            float v = src[i];
            bf16_t h = (bf16_t)v;
            dh[i] = h;
            dl[i] = (bf16_t)(v - (float)h);
        }
    }
    __syncthreads();

    const int wave = t >> 5;
    const int lane = t & 31;
    const int m    = lane & 15;   // row-within-tile for A, column for B/C
    const int half = lane >> 4;

    float bestv[M_TILES][8];
    int   besti[M_TILES][8];
    #pragma unroll
    for (int mt = 0; mt < M_TILES; ++mt)
        #pragma unroll
        for (int r = 0; r < 8; ++r) { bestv[mt][r] = -3.4e38f; besti[mt][r] = 0; }

    const v16bf* pBhi = (const v16bf*)bhi;
    const v16bf* pBlo = (const v16bf*)blo;

    for (int it = 0; it < N_CODES / (16 * 8); ++it) {   // 64 sweep steps
        const int ct = it * 8 + wave;                    // ascending cols/wave
        v8f acc[M_TILES];
        #pragma unroll
        for (int mt = 0; mt < M_TILES; ++mt)
            acc[mt] = (v8f){0.f,0.f,0.f,0.f,0.f,0.f,0.f,0.f};

        #pragma unroll
        for (int s = 0; s < K_STEPS; ++s) {
            const size_t bidx = ((size_t)ct * 8 + s) * 32 + lane;
            v16bf bh = pBhi[bidx];
            v16bf bl = pBlo[bidx];

            v16bf ah[M_TILES], al[M_TILES];
            #pragma unroll
            for (int mt = 0; mt < M_TILES; ++mt) {
                const int off0 = (mt * 16 + m) * AST + s * 32 + half * 8;
                v8bf a0h = *(const v8bf*)(sAhi + off0);
                v8bf a1h = *(const v8bf*)(sAhi + off0 + 16);
                v8bf a0l = *(const v8bf*)(sAlo + off0);
                v8bf a1l = *(const v8bf*)(sAlo + off0 + 16);
                ah[mt] = __builtin_shufflevector(a0h, a1h,
                            0,1,2,3,4,5,6,7,8,9,10,11,12,13,14,15);
                al[mt] = __builtin_shufflevector(a0l, a1l,
                            0,1,2,3,4,5,6,7,8,9,10,11,12,13,14,15);
            }
            // Product-major order: independent accumulators interleave.
            #pragma unroll
            for (int mt = 0; mt < M_TILES; ++mt)
                acc[mt] = __builtin_amdgcn_wmma_f32_16x16x32_bf16(
                            false, ah[mt], false, bh, (short)0, acc[mt], false, false);
            #pragma unroll
            for (int mt = 0; mt < M_TILES; ++mt)
                acc[mt] = __builtin_amdgcn_wmma_f32_16x16x32_bf16(
                            false, ah[mt], false, bl, (short)0, acc[mt], false, false);
            #pragma unroll
            for (int mt = 0; mt < M_TILES; ++mt)
                acc[mt] = __builtin_amdgcn_wmma_f32_16x16x32_bf16(
                            false, al[mt], false, bh, (short)0, acc[mt], false, false);
        }
        // C layout: lane holds column n = ct*16 + m, VGPR r holds row r+8*half
        const int col = ct * 16 + m;
        const float b = bias[col];
        #pragma unroll
        for (int mt = 0; mt < M_TILES; ++mt)
            #pragma unroll
            for (int r = 0; r < 8; ++r) {
                float sc = acc[mt][r] + b;
                if (sc > bestv[mt][r]) { bestv[mt][r] = sc; besti[mt][r] = col; }
            }
    }

    #pragma unroll
    for (int mt = 0; mt < M_TILES; ++mt)
        #pragma unroll
        for (int r = 0; r < 8; ++r) {
            int row = mt * 16 + r + 8 * half;
            r_val[wave][row][m] = bestv[mt][r];
            r_idx[wave][row][m] = besti[mt][r];
        }
    __syncthreads();

    // Cross-wave/lane reduction; prefer smaller index on ties (first argmax).
    if (t < BLK_ROWS) {
        float bv = -3.4e38f; int bi = 0x7fffffff;
        for (int w = 0; w < 8; ++w)
            for (int l = 0; l < 16; ++l) {
                float v = r_val[w][t][l]; int i = r_idx[w][t][l];
                if (v > bv || (v == bv && i < bi)) { bv = v; bi = i; }
            }
        s_best[t] = bi;
        out_i[row0 + t] = (float)bi;
    }
    __syncthreads();

    // Gather quantize output from the ORIGINAL fp32 embed (bit-exact vectors).
    {
        int row8 = t >> 3;             // 0..31
        int c0   = (t & 7) * 32;
        const float* src = embed + (size_t)s_best[row8] * DIM + c0;
        float* dst = out_q + (size_t)(row0 + row8) * DIM + c0;
        #pragma unroll 8
        for (int k = 0; k < 32; ++k) dst[k] = src[k];
    }
}

// ---------------------------------------------------------------------------
extern "C" void kernel_launch(void* const* d_in, const int* in_sizes, int n_in,
                              void* d_out, int out_size, void* d_ws, size_t ws_size,
                              hipStream_t stream) {
    (void)in_sizes; (void)n_in; (void)out_size; (void)ws_size;
    const float* x     = (const float*)d_in[0];
    const float* embed = (const float*)d_in[1];

    char*   ws   = (char*)d_ws;
    bf16_t* bhi  = (bf16_t*)(ws + WS_BHI_OFF);
    bf16_t* blo  = (bf16_t*)(ws + WS_BLO_OFF);
    float*  bias = (float*)(ws + WS_BIAS_OFF);

    float* out_q = (float*)d_out;
    float* out_i = out_q + (size_t)N_ROWS * DIM;

    vq_bias_kernel<<<N_CODES / 256, 256, 0, stream>>>(embed, bias);
    vq_pack_kernel<<<(N_CODES / 16) * 8 * 32 / 256, 256, 0, stream>>>(embed, bhi, blo);
    vq_argmin_kernel<<<N_ROWS / BLK_ROWS, 256, 0, stream>>>(x, embed, bhi, blo,
                                                            bias, out_q, out_i);
}
